// Attention_32839319945390
// MI455X (gfx1250) — compile-verified, matrix operand order checked by reading, not run on hardware
//
#include <hip/hip_runtime.h>
#include <stdint.h>

// ---------------------------------------------------------------------------
// Attention layer for MI455X (gfx1250)
//   - bf16 WMMA (v_wmma_f32_16x16x32_bf16) for all matrix math
//   - Tensor Data Mover (tensor_load_to_lds) double-buffered GEMM tile staging
//   - flash-style streaming-softmax attention (no S x S materialization)
// ---------------------------------------------------------------------------

typedef __bf16 bf16_t;
typedef __bf16 bf16x8  __attribute__((ext_vector_type(8)));
typedef __bf16 bf16x16 __attribute__((ext_vector_type(16)));
typedef float  f32x8   __attribute__((ext_vector_type(8)));
typedef unsigned int u32x4 __attribute__((ext_vector_type(4)));
typedef unsigned int u32x8 __attribute__((ext_vector_type(8)));

#define B_   2
#define S_   2048
#define D_   1024
#define H_   16
#define DH_  64
#define M_   (B_ * S_)        // 4096 rows in the big GEMMs
#define LSTR 40               // padded LDS row stride in bf16 (64B data + 16B pad)

// Pack the two 8-element bf16 runs of a 16-bit A/B fragment:
// per the CDNA5 16-bit 16x32 layout, lane L (row = L&15) holds
// K = k0..k0+7 and K = k0+16..k0+23 where k0 = (L>>4)*8.
__device__ __forceinline__ bf16x16 pack16(const bf16_t* p) {
    bf16x8 lo = *(const bf16x8*)(p);
    bf16x8 hi = *(const bf16x8*)(p + 16);
    bf16x16 f;
#pragma unroll
    for (int i = 0; i < 8; ++i) { f[i] = lo[i]; f[i + 8] = hi[i]; }
    return f;
}

__device__ __forceinline__ bf16x16 frag_from(const bf16_t* rowbase, int stride, int lane) {
    int r  = lane & 15;
    int k0 = (lane >> 4) << 3;
    return pack16(rowbase + (size_t)r * stride + k0);
}

// Low 32 bits of a generic pointer to __shared__ = workgroup-relative LDS byte
// address (aperture encoding: addr[31:0] is the LDS offset).
__device__ __forceinline__ unsigned lds_addr32(const void* p) {
    return (unsigned)(uintptr_t)p;
}

// ---------------------------------------------------------------------------
// TDM: issue a 2-D tile load (tile_cols=32 bf16 x tile_rows=128) from global
// into LDS with 16B padding after every 64B row => LDS row stride = 80B (LSTR).
// D# layout per CDNA5 ISA ch.8 (group0: 4 SGPRs, group1: 8 SGPRs).
// Tracked by TENSORcnt; TDM ignores EXEC, caller must gate to one wave.
// ---------------------------------------------------------------------------
__device__ __forceinline__ void tdm_load_tile(unsigned lds_off, const bf16_t* gptr,
                                              unsigned kdim) {
    unsigned long long ga = (unsigned long long)(uintptr_t)gptr;
    u32x4 g0;
    g0[0] = 1u;                                           // count=1, user descriptor
    g0[1] = lds_off;                                      // LDS byte address
    g0[2] = (unsigned)ga;                                 // global addr [31:0]
    g0[3] = (unsigned)((ga >> 32) & 0x1FFFFFFu)           // global addr [56:32]
          | (2u << 30);                                   // type = 2 ("image")
    u32x8 g1;
    g1[0] = (1u << 16)                                    // data_size = 2 bytes
          | (1u << 20)                                    // pad_enable
          | (3u << 22)                                    // pad_interval: every 64B
          | (3u << 25);                                   // pad_amount: 4 DWORDs (16B)
    g1[1] = (kdim & 0xFFFFu) << 16;                       // tensor_dim0[15:0]
    g1[2] = (kdim >> 16) & 0xFFFFu;                       // tensor_dim0[31:16], dim1 lo=0
    g1[3] = 16u | (32u << 16);                            // tensor_dim1=1<<20, tile_dim0=32
    g1[4] = 128u;                                         // tile_dim1=128, tile_dim2=0
    g1[5] = kdim;                                         // tensor_dim0_stride[31:0]
    g1[6] = 0u;                                           // stride0 hi, stride1 lo
    g1[7] = 0u;                                           // stride1 hi
    asm volatile("tensor_load_to_lds %0, %1" :: "s"(g0), "s"(g1) : "memory");
}

// ---------------------------------------------------------------------------
// f32 -> bf16 elementwise
// ---------------------------------------------------------------------------
__global__ __launch_bounds__(256) void k_cvt(const float* __restrict__ x,
                                             bf16_t* __restrict__ y, int n) {
    int i = blockIdx.x * 256 + threadIdx.x;
    if (i < n) y[i] = (bf16_t)x[i];
}

// ---------------------------------------------------------------------------
// W[k][n] (f32, DxD) -> Wt[n][k] (bf16)  so GEMM B-fragments load row-major
// ---------------------------------------------------------------------------
__global__ __launch_bounds__(256) void k_transpose(const float* __restrict__ W,
                                                   bf16_t* __restrict__ Wt) {
    __shared__ float t[32][33];
    int k0 = blockIdx.x * 32, n0 = blockIdx.y * 32;
    int tx = threadIdx.x & 31, ty = threadIdx.x >> 5;   // ty in 0..7
#pragma unroll
    for (int r = ty; r < 32; r += 8)
        t[r][tx] = W[(size_t)(k0 + r) * D_ + n0 + tx];
    __syncthreads();
#pragma unroll
    for (int r = ty; r < 32; r += 8)
        Wt[(size_t)(n0 + r) * D_ + k0 + tx] = (bf16_t)t[tx][r];
}

// ---------------------------------------------------------------------------
// C[m][n] = sum_k A[m][k] * Bt[n][k] + bias[n]
//   A : M x K bf16 row-major,  Bt : N x K bf16 row-major (pre-transposed W)
//   mode 0: store bf16 head-split  [B,H,S,DH]   (Q and K projections)
//   mode 1: store bf16 head-split-T [B,H,DH,S]  (V projection, for P@V frags)
//   mode 2: store f32 row-major M x D           (final output)
// Block: 256 threads (8 waves, 2x4), tile 128x128, BK=32.
// Tiles staged into LDS by the Tensor Data Mover, double buffered: wave 0
// issues the next A/B tile pair, waits TENSORcnt<=2 (current pair landed),
// then all waves barrier and run 8 WMMAs per step.
// ---------------------------------------------------------------------------
__global__ __launch_bounds__(256) void k_gemm(const bf16_t* __restrict__ A,
                                              const bf16_t* __restrict__ Bt,
                                              const float* __restrict__ bias,
                                              bf16_t* __restrict__ obf,
                                              float* __restrict__ of32,
                                              int mode, int Kdim) {
    __shared__ __align__(16) bf16_t As[2][128 * LSTR];
    __shared__ __align__(16) bf16_t Bs[2][128 * LSTR];

    int tid = threadIdx.x, lane = tid & 31, wid = tid >> 5;
    int wm = wid & 1, wn = wid >> 1;                 // 2 waves over M, 4 over N
    int bm0 = blockIdx.x * 128, bn0 = blockIdx.y * 128;
    // provably-scalar wave id so the TDM branch is a uniform s_cbranch
    // (TDM ignores EXEC; must not be reached by waves with EXEC=0)
    int wsel = __builtin_amdgcn_readfirstlane(tid) >> 5;

    const f32x8 zero8 = {0.f, 0.f, 0.f, 0.f, 0.f, 0.f, 0.f, 0.f};
    f32x8 acc[4][2];
#pragma unroll
    for (int m = 0; m < 4; ++m)
#pragma unroll
        for (int n = 0; n < 2; ++n) acc[m][n] = zero8;

    const bf16_t* Abase = A  + (size_t)bm0 * Kdim;
    const bf16_t* Bbase = Bt + (size_t)bn0 * Kdim;
    int nk = Kdim >> 5;

    if (wsel == 0) {
        tdm_load_tile(lds_addr32(&As[0][0]), Abase, (unsigned)Kdim);
        tdm_load_tile(lds_addr32(&Bs[0][0]), Bbase, (unsigned)Kdim);
    }

    for (int i = 0; i < nk; ++i) {
        int cur = i & 1;
        if (wsel == 0) {
            if (i + 1 < nk) {
                tdm_load_tile(lds_addr32(&As[cur ^ 1][0]), Abase + (i + 1) * 32,
                              (unsigned)Kdim);
                tdm_load_tile(lds_addr32(&Bs[cur ^ 1][0]), Bbase + (i + 1) * 32,
                              (unsigned)Kdim);
                __builtin_amdgcn_s_wait_tensorcnt(2);   // current pair complete
            } else {
                __builtin_amdgcn_s_wait_tensorcnt(0);
            }
        }
        __syncthreads();                                // publish buffer `cur`

        bf16x16 af[4], bfr[2];
#pragma unroll
        for (int m = 0; m < 4; ++m)
            af[m] = frag_from(&As[cur][(wm * 64 + m * 16) * LSTR], LSTR, lane);
#pragma unroll
        for (int n = 0; n < 2; ++n)
            bfr[n] = frag_from(&Bs[cur][(wn * 32 + n * 16) * LSTR], LSTR, lane);

#pragma unroll
        for (int m = 0; m < 4; ++m)
#pragma unroll
            for (int n = 0; n < 2; ++n)
                acc[m][n] = __builtin_amdgcn_wmma_f32_16x16x32_bf16(
                    false, af[m], false, bfr[n], (short)0, acc[m][n], false, false);
        __syncthreads();                                // done reading before overwrite
    }

    // Epilogue: C-fragment layout -> lanes 0-15 hold M=v, lanes 16-31 hold M=v+8.
    int rsel = (lane >> 4) << 3;
#pragma unroll
    for (int m = 0; m < 4; ++m) {
#pragma unroll
        for (int n = 0; n < 2; ++n) {
#pragma unroll
            for (int v = 0; v < 8; ++v) {
                int row = bm0 + wm * 64 + m * 16 + rsel + v;
                int col = bn0 + wn * 32 + n * 16 + (lane & 15);
                float val = acc[m][n][v] + bias[col];
                if (mode == 2) {
                    of32[(size_t)row * D_ + col] = val;
                } else {
                    int b = row >> 11, s = row & (S_ - 1);
                    int h = col >> 6,  dh = col & (DH_ - 1);
                    size_t idx = (mode == 0)
                        ? ((size_t)(b * H_ + h) * S_ + s) * DH_ + dh
                        : ((size_t)(b * H_ + h) * DH_ + dh) * S_ + s;
                    obf[idx] = (bf16_t)val;
                }
            }
        }
    }
}

// ---------------------------------------------------------------------------
// Flash attention: grid (S/64, B*H), 128 threads = 4 waves, 16 q-rows / wave.
// Streams keys in tiles of 32; online softmax with row stats in VGPRs; row
// reductions via half-wave butterfly shuffles; P re-layout via per-wave LDS.
// Fuses residual (+ q_) and bf16 conversion for the final GEMM input.
// ---------------------------------------------------------------------------
__global__ __launch_bounds__(128) void k_flash(const bf16_t* __restrict__ Qh,
                                               const bf16_t* __restrict__ Kh,
                                               const bf16_t* __restrict__ Vt,
                                               const unsigned char* __restrict__ mask,
                                               const float* __restrict__ qres,
                                               bf16_t* __restrict__ Xf) {
    __shared__ __align__(16) bf16_t Pb[4][16 * LSTR];   // per-wave 16x32 P tile

    int tid = threadIdx.x, lane = tid & 31, wid = tid >> 5;
    int bh = blockIdx.y, b = bh >> 4, h = bh & 15;
    int q16 = blockIdx.x * 64 + wid * 16;

    const bf16_t* Qb = Qh + (size_t)bh * S_ * DH_;
    const bf16_t* Kb = Kh + (size_t)bh * S_ * DH_;
    const bf16_t* Vb = Vt + (size_t)bh * DH_ * S_;

    int rsel = (lane >> 4) << 3;

    // Q fragments (DH=64 -> two K=32 chunks), loaded once.
    bf16x16 aq[2];
#pragma unroll
    for (int c = 0; c < 2; ++c)
        aq[c] = frag_from(Qb + (size_t)q16 * DH_ + c * 32, DH_, lane);

    // Row mask (reference masks whole query rows, broadcast over keys).
    bool mok[8];
#pragma unroll
    for (int v = 0; v < 8; ++v)
        mok[v] = mask[(size_t)b * S_ + q16 + rsel + v] != 0;

    const f32x8 zero8 = {0.f, 0.f, 0.f, 0.f, 0.f, 0.f, 0.f, 0.f};
    float mrow[8], lrow[8];
    f32x8 o[4];
#pragma unroll
    for (int v = 0; v < 8; ++v) { mrow[v] = -3.0e38f; lrow[v] = 0.f; }
#pragma unroll
    for (int d = 0; d < 4; ++d) o[d] = zero8;

    for (int kt = 0; kt < S_; kt += 32) {
        // scores: C0 = keys kt..kt+15, C1 = keys kt+16..kt+31
        f32x8 c0 = zero8, c1 = zero8;
#pragma unroll
        for (int c = 0; c < 2; ++c) {
            bf16x16 bk0 = frag_from(Kb + (size_t)kt * DH_ + c * 32, DH_, lane);
            bf16x16 bk1 = frag_from(Kb + (size_t)(kt + 16) * DH_ + c * 32, DH_, lane);
            c0 = __builtin_amdgcn_wmma_f32_16x16x32_bf16(false, aq[c], false, bk0,
                                                         (short)0, c0, false, false);
            c1 = __builtin_amdgcn_wmma_f32_16x16x32_bf16(false, aq[c], false, bk1,
                                                         (short)0, c1, false, false);
        }

        float p0[8], p1[8], rmax[8], rsum[8];
#pragma unroll
        for (int v = 0; v < 8; ++v) {
            p0[v] = mok[v] ? c0[v] * 0.125f : -1.0e12f;   // 1/sqrt(64)
            p1[v] = mok[v] ? c1[v] * 0.125f : -1.0e12f;
            rmax[v] = fmaxf(p0[v], p1[v]);
        }
        // row max across the 16 lanes of a half-wave
#pragma unroll
        for (int off = 1; off < 16; off <<= 1)
#pragma unroll
            for (int v = 0; v < 8; ++v)
                rmax[v] = fmaxf(rmax[v], __shfl_xor(rmax[v], off, 32));

#pragma unroll
        for (int v = 0; v < 8; ++v) {
            float mn = fmaxf(mrow[v], rmax[v]);
            float al = __expf(mrow[v] - mn);
            mrow[v] = mn;
            p0[v] = __expf(p0[v] - mn);
            p1[v] = __expf(p1[v] - mn);
            rsum[v] = p0[v] + p1[v];
            lrow[v] *= al;
#pragma unroll
            for (int d = 0; d < 4; ++d) o[d][v] *= al;
            // stash P tile (C-fragment -> row-major LDS) for A-fragment reload
            Pb[wid][(rsel + v) * LSTR + (lane & 15)]      = (bf16_t)p0[v];
            Pb[wid][(rsel + v) * LSTR + 16 + (lane & 15)] = (bf16_t)p1[v];
        }
#pragma unroll
        for (int off = 1; off < 16; off <<= 1)
#pragma unroll
            for (int v = 0; v < 8; ++v)
                rsum[v] += __shfl_xor(rsum[v], off, 32);
#pragma unroll
        for (int v = 0; v < 8; ++v) lrow[v] += rsum[v];

        // intra-wave LDS RAW: DS ops are in-order per wave; wait to be safe
        asm volatile("s_wait_dscnt 0" ::: "memory");

        bf16x16 pf = frag_from(&Pb[wid][0], LSTR, lane);
#pragma unroll
        for (int d = 0; d < 4; ++d) {
            bf16x16 bv = frag_from(Vb + (size_t)(d * 16) * S_ + kt, S_, lane);
            o[d] = __builtin_amdgcn_wmma_f32_16x16x32_bf16(false, pf, false, bv,
                                                           (short)0, o[d], false, false);
        }
    }

    // Epilogue: normalize, add residual q_, emit bf16 for the final GEMM.
#pragma unroll
    for (int d = 0; d < 4; ++d) {
#pragma unroll
        for (int v = 0; v < 8; ++v) {
            int row = q16 + rsel + v;
            int col = h * DH_ + d * 16 + (lane & 15);
            size_t gi = ((size_t)b * S_ + row) * D_ + col;
            float val = o[d][v] / lrow[v];
            Xf[gi] = (bf16_t)(val + qres[gi]);
        }
    }
}

// ---------------------------------------------------------------------------
// Host launcher
// ---------------------------------------------------------------------------
extern "C" void kernel_launch(void* const* d_in, const int* in_sizes, int n_in,
                              void* d_out, int out_size, void* d_ws, size_t ws_size,
                              hipStream_t stream) {
    const float* q_ = (const float*)d_in[0];
    const float* k_ = (const float*)d_in[1];
    const float* v_ = (const float*)d_in[2];
    const unsigned char* mask = (const unsigned char*)d_in[3];
    const float* Wq = (const float*)d_in[4];
    const float* bq = (const float*)d_in[5];
    const float* Wk = (const float*)d_in[6];
    const float* bk = (const float*)d_in[7];
    const float* Wv = (const float*)d_in[8];
    const float* bv = (const float*)d_in[9];
    const float* Wf = (const float*)d_in[10];
    const float* bf = (const float*)d_in[11];
    float* out = (float*)d_out;

    char*  ws = (char*)d_ws;
    size_t off = 0;
    auto alloc = [&](size_t bytes) -> void* {
        void* p = ws + off;
        off += (bytes + 255) & ~(size_t)255;
        return p;
    };
    const size_t WB = (size_t)D_ * D_ * sizeof(bf16_t);        // 2 MB
    const size_t XB = (size_t)B_ * S_ * D_ * sizeof(bf16_t);   // 8 MB
    bf16_t* WqT = (bf16_t*)alloc(WB);
    bf16_t* WkT = (bf16_t*)alloc(WB);
    bf16_t* WvT = (bf16_t*)alloc(WB);
    bf16_t* WfT = (bf16_t*)alloc(WB);
    bf16_t* Xq  = (bf16_t*)alloc(XB);
    bf16_t* Xk  = (bf16_t*)alloc(XB);
    bf16_t* Xv  = (bf16_t*)alloc(XB);
    bf16_t* Qh  = (bf16_t*)alloc(XB);
    bf16_t* Kh  = (bf16_t*)alloc(XB);
    bf16_t* Vt  = (bf16_t*)alloc(XB);
    bf16_t* Xf  = (bf16_t*)alloc(XB);

    // Stage weights (transposed) and activations as bf16.
    dim3 gT(D_ / 32, D_ / 32);
    k_transpose<<<gT, 256, 0, stream>>>(Wq, WqT);
    k_transpose<<<gT, 256, 0, stream>>>(Wk, WkT);
    k_transpose<<<gT, 256, 0, stream>>>(Wv, WvT);
    k_transpose<<<gT, 256, 0, stream>>>(Wf, WfT);
    int n = B_ * S_ * D_;
    k_cvt<<<n / 256, 256, 0, stream>>>(q_, Xq, n);
    k_cvt<<<n / 256, 256, 0, stream>>>(k_, Xk, n);
    k_cvt<<<n / 256, 256, 0, stream>>>(v_, Xv, n);

    // QKV projections (fused bias + head-split layouts).
    dim3 gg(M_ / 128, D_ / 128);
    k_gemm<<<gg, 256, 0, stream>>>(Xq, WqT, bq, Qh, nullptr, 0, D_);
    k_gemm<<<gg, 256, 0, stream>>>(Xk, WkT, bk, Kh, nullptr, 0, D_);
    k_gemm<<<gg, 256, 0, stream>>>(Xv, WvT, bv, Vt, nullptr, 1, D_);

    // Flash attention + residual fuse.
    dim3 gf(S_ / 64, B_ * H_);
    k_flash<<<gf, 128, 0, stream>>>(Qh, Kh, Vt, mask, q_, Xf);

    // Final projection -> f32 output.
    k_gemm<<<gg, 256, 0, stream>>>(Xf, WfT, bf, nullptr, out, 2, D_);
}
